// EncoderLayer_22239340658894
// MI455X (gfx1250) — matrix operand order repaired and checked by reading, not compiled, not run on hardware
//
#include <hip/hip_runtime.h>
#include <hip/hip_bf16.h>

// ---------------------------------------------------------------------------
// Transformer encoder layer for MI455X (gfx1250, wave32, WMMA bf16).
// B=4, S=2048, D=1024, H=16, DH=64.  All matrix math on v_wmma_f32_16x16x32_bf16.
// GEMM tiles staged with CDNA5 async global->LDS DMA (ASYNCcnt) when the
// toolchain exposes the builtins; attention keeps softmax fully in registers.
// ---------------------------------------------------------------------------

typedef __bf16 bf16;
typedef bf16  v16bf __attribute__((ext_vector_type(16)));
typedef bf16  v8bf  __attribute__((ext_vector_type(8)));
typedef bf16  v4bf  __attribute__((ext_vector_type(4)));
typedef float v8f   __attribute__((ext_vector_type(8)));
typedef int   v4i   __attribute__((vector_size(16)));  // b128 payload type

#define WMMA_BF16(a, b, c)                                                    \
  __builtin_amdgcn_wmma_f32_16x16x32_bf16(false, (a), false, (b), (short)0,   \
                                          (c), false, false)

#if __has_builtin(__builtin_amdgcn_global_load_async_to_lds_b128) && \
    __has_builtin(__builtin_amdgcn_s_wait_asynccnt)
#define HAVE_ASYNC_LDS 1
// builtin signature (from probe): (as1 v4i* gsrc, as3 v4i* lds, imm off, imm cpol)
#define AS1CAST(p) ((__attribute__((address_space(1))) v4i*)(p))
#define AS3CAST(p) ((__attribute__((address_space(3))) v4i*)(p))
#endif

static constexpr int BDIM = 4;
static constexpr int SDIM = 2048;
static constexpr int DDIM = 1024;
static constexpr int HDIM = 16;
static constexpr int DH   = 64;
static constexpr int MDIM = BDIM * SDIM;  // 8192 rows

// ---------------------------------------------------------------------------
// f32 -> bf16 conversion (vectorized x4)
// ---------------------------------------------------------------------------
__global__ __launch_bounds__(256) void cvt_bf16_kernel(
    const float* __restrict__ in, bf16* __restrict__ out, int n4) {
  int i = blockIdx.x * 256 + threadIdx.x;
  if (i < n4) {
    float4 f = ((const float4*)in)[i];
    v4bf o = {(bf16)f.x, (bf16)f.y, (bf16)f.z, (bf16)f.w};
    ((v4bf*)out)[i] = o;
  }
}

// ---------------------------------------------------------------------------
// GEMM: C[M,N] = A[M,K](bf16) @ W[K,N](bf16) + bias, templated epilogue.
//   MODE 0: bf16 out in [B,H,S,DH]   (K, V row-major per head)
//   MODE 1: bf16 out in [B,H,DH,S]   (Q transposed: B-operand of scores WMMA)
//   MODE 3: f32 out, plain [M,N]     (attention output projection)
//   MODE 4: f32 out, ReLU            (FFN)
// Workgroup: 256 threads (8 waves) computes 128x128; each wave 32(M) x 64(N)
// = 8 WMMA tiles.  LDS double-buffered; tiles staged with async global->LDS
// DMA (no VGPR staging) when available.
// ---------------------------------------------------------------------------
template <int MODE>
__global__ __launch_bounds__(256) void gemm_bf16_kernel(
    const bf16* __restrict__ A, const bf16* __restrict__ W,
    const float* __restrict__ bias, void* __restrict__ out, int K, int N) {
  constexpr int BM = 128, BN = 128, BK = 32;
  __shared__ bf16 sA[2][BM][BK + 8];  // stride 40 halves (80B, 16B aligned)
  __shared__ bf16 sW[2][BK][BN + 8];  // stride 136 halves (272B, 16B aligned)

  const int tid  = threadIdx.x;
  const int wave = tid >> 5;
  const int lane = tid & 31;
  const int lg   = lane >> 4;   // lane group (0: lanes 0-15, 1: lanes 16-31)
  const int lr   = lane & 15;
  const int kb8  = lg * 8;      // A-fragment k base per ISA layout

  const int m0 = blockIdx.y * BM;
  const int n0 = blockIdx.x * BN;
  const int wm = (wave >> 1) * 32;  // 0,32,64,96
  const int wn = (wave & 1) * 64;   // 0,64

  // staging-chunk coordinates (8-half = 16B chunks)
  int arow[2], acol[2], wrow[2], wcol[2];
#pragma unroll
  for (int j = 0; j < 2; ++j) {
    int c   = tid + 256 * j;         // A tile: 128x32 = 512 chunks
    arow[j] = c >> 2;
    acol[j] = (c & 3) * 8;
    wrow[j] = c >> 4;                // W tile: 32x128 = 512 chunks
    wcol[j] = (c & 15) * 8;
  }

  const int nit = K / BK;
  v8f acc[2][4] = {};

#ifdef HAVE_ASYNC_LDS
  // -------- async DMA staging: global -> LDS directly (ASYNCcnt) --------
  auto issue_tile = [&](int k0, int buf) {
#pragma unroll
    for (int j = 0; j < 2; ++j)
      __builtin_amdgcn_global_load_async_to_lds_b128(
          AS1CAST(&A[(size_t)(m0 + arow[j]) * K + k0 + acol[j]]),
          AS3CAST(&sA[buf][arow[j]][acol[j]]), 0, 0);
#pragma unroll
    for (int j = 0; j < 2; ++j)
      __builtin_amdgcn_global_load_async_to_lds_b128(
          AS1CAST(&W[(size_t)(k0 + wrow[j]) * N + n0 + wcol[j]]),
          AS3CAST(&sW[buf][wrow[j]][wcol[j]]), 0, 0);
  };
  issue_tile(0, 0);
  __builtin_amdgcn_s_wait_asynccnt(0);
  __syncthreads();
#else
  // -------- fallback: stage through VGPRs --------
  v8bf ra[2], rw[2];
  auto load_tile = [&](int k0) {
#pragma unroll
    for (int j = 0; j < 2; ++j)
      ra[j] = *(const v8bf*)&A[(size_t)(m0 + arow[j]) * K + k0 + acol[j]];
#pragma unroll
    for (int j = 0; j < 2; ++j)
      rw[j] = *(const v8bf*)&W[(size_t)(k0 + wrow[j]) * N + n0 + wcol[j]];
  };
  auto store_tile = [&](int buf) {
#pragma unroll
    for (int j = 0; j < 2; ++j) *(v8bf*)&sA[buf][arow[j]][acol[j]] = ra[j];
#pragma unroll
    for (int j = 0; j < 2; ++j) *(v8bf*)&sW[buf][wrow[j]][wcol[j]] = rw[j];
  };
  load_tile(0);
  store_tile(0);
  __syncthreads();
#endif

  for (int it = 0; it < nit; ++it) {
    const int cur = it & 1;
#ifdef HAVE_ASYNC_LDS
    if (it + 1 < nit) issue_tile((it + 1) * BK, cur ^ 1);
#else
    if (it + 1 < nit) load_tile((it + 1) * BK);
#endif
    if (it + 2 < nit)  // gfx1250 global_prefetch two tiles ahead
      __builtin_prefetch(&A[(size_t)(m0 + (tid >> 1)) * K + (it + 2) * BK], 0, 0);

    // ---- build fragments from current LDS buffers ----
    v16bf af[2], wf[4];
#pragma unroll
    for (int mi = 0; mi < 2; ++mi) {
      int row = wm + mi * 16 + lr;
      v8bf lo = *(const v8bf*)&sA[cur][row][kb8];
      v8bf hi = *(const v8bf*)&sA[cur][row][16 + kb8];
#pragma unroll
      for (int i = 0; i < 8; ++i) { af[mi][i] = lo[i]; af[mi][i + 8] = hi[i]; }
    }
#pragma unroll
    for (int ni = 0; ni < 4; ++ni) {
      int col = wn + ni * 16 + lr;
#pragma unroll
      for (int i = 0; i < 8; ++i) { wf[ni][i] = sW[cur][kb8 + i][col]; wf[ni][i + 8] = sW[cur][16 + kb8 + i][col]; }
    }

#pragma unroll
    for (int mi = 0; mi < 2; ++mi)
#pragma unroll
      for (int ni = 0; ni < 4; ++ni)
        acc[mi][ni] = WMMA_BF16(af[mi], wf[ni], acc[mi][ni]);

#ifdef HAVE_ASYNC_LDS
    if (it + 1 < nit) __builtin_amdgcn_s_wait_asynccnt(0);
#else
    if (it + 1 < nit) store_tile(cur ^ 1);
#endif
    __syncthreads();
  }

  // ---- epilogue: C/D layout => lane gives N=lr, VGPR r gives M=r+8*lg ----
#pragma unroll
  for (int mi = 0; mi < 2; ++mi)
#pragma unroll
    for (int ni = 0; ni < 4; ++ni) {
#pragma unroll
      for (int r = 0; r < 8; ++r) {
        int m = m0 + wm + mi * 16 + r + 8 * lg;
        int n = n0 + wn + ni * 16 + lr;
        float v = acc[mi][ni][r] + bias[n];
        if (MODE == 0) {  // bf16 [B,H,S,DH]
          int b = m >> 11, s = m & 2047, h = n >> 6, dh = n & 63;
          ((bf16*)out)[((size_t)((b * HDIM + h) * SDIM + s)) * DH + dh] =
              (bf16)v;
        } else if (MODE == 1) {  // bf16 [B,H,DH,S]
          int b = m >> 11, s = m & 2047, h = n >> 6, dh = n & 63;
          ((bf16*)out)[((size_t)((b * HDIM + h) * DH + dh)) * SDIM + s] =
              (bf16)v;
        } else if (MODE == 3) {  // f32 plain
          ((float*)out)[(size_t)m * N + n] = v;
        } else {  // f32 relu
          ((float*)out)[(size_t)m * N + n] = v > 0.f ? v : 0.f;
        }
      }
    }
}

// ---------------------------------------------------------------------------
// Flash attention (transposed scores): one wave owns a 16-query block.
//   Qt:[B,H,DH,S] bf16 (transposed), K:[B,H,S,DH], V:[B,H,S,DH].
// Scores tile computed as S^T = K_blk @ Q^T, so the D-layout puts keys in the
// register dim and queries in lanes:
//   - softmax max/sum: in-register + ONE __shfl_xor(16)
//   - P in S^T D-layout IS the A-fragment layout for P@V: element i of lane L
//     needs key (i&7)+(i<8?0:16)+8*(L/16) for query L%16 == st[i>=8][i&7].
// ctx written to [B,S,D] bf16 for the output-projection GEMM.
// ---------------------------------------------------------------------------
__global__ __launch_bounds__(256) void flash_attn_kernel(
    const bf16* __restrict__ Qt, const bf16* __restrict__ K,
    const bf16* __restrict__ V, bf16* __restrict__ ctx) {
  const int tid  = threadIdx.x;
  const int wave = tid >> 5;
  const int lane = tid & 31;
  const int lg   = lane >> 4;
  const int lr   = lane & 15;
  const int kb8  = lg * 8;

  const int task = blockIdx.x * 8 + wave;  // 8192 tasks = B*H*(S/16)
  const int bh   = task >> 7;
  const int qb   = task & 127;
  const int q0   = qb * 16;

  const bf16* Qtp = Qt + (size_t)bh * DH * SDIM;
  const bf16* Kp  = K + (size_t)bh * SDIM * DH;
  const bf16* Vp  = V + (size_t)bh * SDIM * DH;

  // B-operand Q fragments (held all loop): lane = query q0+lr,
  // 16 d values in table k order.  Pre-scale by 1/sqrt(DH)=0.125 (exact in bf16).
  v16bf qf[2];
#pragma unroll
  for (int ds = 0; ds < 2; ++ds) {
#pragma unroll
    for (int i = 0; i < 16; ++i) {
      int d = ds * 32 + (i < 8 ? kb8 + i : 8 + kb8 + i);
      qf[ds][i] = (bf16)((float)Qtp[(size_t)d * SDIM + q0 + lr] * 0.125f);
    }
  }

  float mq = -1e30f, lq = 0.f;  // per-lane: stats of query q = lr
  v8f acc[4] = {};

  for (int kb = 0; kb < SDIM; kb += 32) {
    // ---- S^T tiles: D[key][q]; A-operand = K rows ----
    v8f st[2];
#pragma unroll
    for (int kk = 0; kk < 2; ++kk) {
      const bf16* rowp = Kp + (size_t)(kb + kk * 16 + lr) * DH;
      v8f s = {};
#pragma unroll
      for (int ds = 0; ds < 2; ++ds) {
        v8bf lo = *(const v8bf*)(rowp + ds * 32 + kb8);
        v8bf hi = *(const v8bf*)(rowp + ds * 32 + 16 + kb8);
        v16bf af;
#pragma unroll
        for (int i = 0; i < 8; ++i) { af[i] = lo[i]; af[i + 8] = hi[i]; }
        s = WMMA_BF16(af, qf[ds], s);
      }
      st[kk] = s;
    }

    // ---- online softmax, all in registers (+1 cross-lane combine) ----
    float vmax = st[0][0];
#pragma unroll
    for (int r = 1; r < 8; ++r) vmax = fmaxf(vmax, st[0][r]);
#pragma unroll
    for (int r = 0; r < 8; ++r) vmax = fmaxf(vmax, st[1][r]);
    vmax = fmaxf(vmax, __shfl_xor(vmax, 16, 32));

    const float mnew = fmaxf(mq, vmax);
    const float corr = __expf(mq - mnew);
    mq = mnew;

    v16bf pf;  // = A-fragment of P for the PV WMMA (register repack only)
    float tsum = 0.f;
#pragma unroll
    for (int kk = 0; kk < 2; ++kk)
#pragma unroll
      for (int r = 0; r < 8; ++r) {
        float p = __expf(st[kk][r] - mnew);
        tsum += p;
        pf[kk * 8 + r] = (bf16)p;
      }
    tsum += __shfl_xor(tsum, 16, 32);
    lq = lq * corr + tsum;

    // rescale acc: acc rows are queries r+8*lg; broadcast corr from lane q
    float cr[8];
#pragma unroll
    for (int r = 0; r < 8; ++r) cr[r] = __shfl(corr, r + 8 * lg, 32);
#pragma unroll
    for (int t = 0; t < 4; ++t)
#pragma unroll
      for (int r = 0; r < 8; ++r) acc[t][r] *= cr[r];

    // ---- P @ V: B-operand lane holds V column d=t*16+lr, 16 keys in table order ----
#pragma unroll
    for (int t = 0; t < 4; ++t) {
      v16bf vf;
#pragma unroll
      for (int i = 0; i < 16; ++i)
        vf[i] = Vp[(size_t)(kb + (i < 8 ? kb8 + i : 8 + kb8 + i)) * DH + t * 16 + lr];
      acc[t] = WMMA_BF16(pf, vf, acc[t]);
    }
  }

  // ---- normalize (broadcast 1/l per register row) and write ctx [B,S,D] ----
  const float inv = 1.f / lq;
  float ir[8];
#pragma unroll
  for (int r = 0; r < 8; ++r) ir[r] = __shfl(inv, r + 8 * lg, 32);

  const int b = bh >> 4, h = bh & 15;
#pragma unroll
  for (int t = 0; t < 4; ++t)
#pragma unroll
    for (int r = 0; r < 8; ++r) {
      int m = q0 + r + 8 * lg;
      int n = h * DH + t * 16 + lr;
      ctx[(size_t)(b * SDIM + m) * DDIM + n] = (bf16)(acc[t][r] * ir[r]);
    }
}

// ---------------------------------------------------------------------------
// out = LayerNorm(A + B) * gamma + beta; optionally also emit bf16 copy.
// One workgroup per row (D=1024, 4 elems/thread).
// ---------------------------------------------------------------------------
template <bool WRITE_BF16>
__global__ __launch_bounds__(256) void add_layernorm_kernel(
    const float* __restrict__ A, const float* __restrict__ Bv,
    const float* __restrict__ g, const float* __restrict__ be,
    float* __restrict__ outf, bf16* __restrict__ outb) {
  __shared__ float red[256];
  const int row = blockIdx.x;
  const int tid = threadIdx.x;
  const float* a = A + (size_t)row * DDIM;
  const float* b = Bv + (size_t)row * DDIM;

  float v[4];
  float s = 0.f;
#pragma unroll
  for (int j = 0; j < 4; ++j) {
    int i = tid + j * 256;
    v[j] = a[i] + b[i];
    s += v[j];
  }
  red[tid] = s;
  __syncthreads();
  for (int st = 128; st > 0; st >>= 1) {
    if (tid < st) red[tid] += red[tid + st];
    __syncthreads();
  }
  const float mu = red[0] * (1.f / DDIM);
  __syncthreads();

  float s2 = 0.f;
#pragma unroll
  for (int j = 0; j < 4; ++j) {
    float d = v[j] - mu;
    s2 += d * d;
  }
  red[tid] = s2;
  __syncthreads();
  for (int st = 128; st > 0; st >>= 1) {
    if (tid < st) red[tid] += red[tid + st];
    __syncthreads();
  }
  const float rstd = rsqrtf(red[0] * (1.f / DDIM) + 1e-6f);

#pragma unroll
  for (int j = 0; j < 4; ++j) {
    int i = tid + j * 256;
    float o = (v[j] - mu) * rstd * g[i] + be[i];
    outf[(size_t)row * DDIM + i] = o;
    if (WRITE_BF16) outb[(size_t)row * DDIM + i] = (bf16)o;
  }
}

// ---------------------------------------------------------------------------
// Host-side orchestration
// ---------------------------------------------------------------------------
extern "C" void kernel_launch(void* const* d_in, const int* in_sizes, int n_in,
                              void* d_out, int out_size, void* d_ws,
                              size_t ws_size, hipStream_t stream) {
  (void)in_sizes; (void)n_in; (void)out_size; (void)ws_size;

  const float* x  = (const float*)d_in[0];
  const float* wq = (const float*)d_in[1];
  const float* bq = (const float*)d_in[2];
  const float* wk = (const float*)d_in[3];
  const float* bk = (const float*)d_in[4];
  const float* wv = (const float*)d_in[5];
  const float* bv = (const float*)d_in[6];
  const float* wo = (const float*)d_in[7];
  const float* bo = (const float*)d_in[8];
  const float* wf = (const float*)d_in[9];
  const float* bff = (const float*)d_in[10];
  const float* g1 = (const float*)d_in[11];
  const float* b1 = (const float*)d_in[12];
  const float* g2 = (const float*)d_in[13];
  const float* b2 = (const float*)d_in[14];
  float* out = (float*)d_out;

  const size_t MD = (size_t)MDIM * DDIM;  // 8,388,608 elems
  const size_t DD = (size_t)DDIM * DDIM;

  // Workspace carve (aliased where lifetimes permit)
  char* w = (char*)d_ws;
  auto alloc = [&](size_t bytes) -> char* {
    char* p = w;
    w += (bytes + 255) & ~(size_t)255;
    return p;
  };
  bf16* xb   = (bf16*)alloc(MD * 2);
  bf16* wqb  = (bf16*)alloc(DD * 2);
  bf16* wkb  = (bf16*)alloc(DD * 2);
  bf16* wvb  = (bf16*)alloc(DD * 2);
  bf16* wob  = (bf16*)alloc(DD * 2);
  bf16* wfb  = (bf16*)alloc(DD * 2);
  bf16* Qtb  = (bf16*)alloc(MD * 2);   // Q transposed [B,H,DH,S]
  bf16* Kb   = (bf16*)alloc(MD * 2);   // K row-major  [B,H,S,DH]
  bf16* Vb   = (bf16*)alloc(MD * 2);   // V row-major  [B,H,S,DH]
  float* y   = (float*)alloc(MD * 4);  // attn_out
  float* n1  = (float*)alloc(MD * 4);
  bf16* ctxb = xb;                     // xb dead after QKV GEMMs
  bf16* n1b  = Qtb;                    // Qtb dead after attention
  float* fn  = y;                      // y dead after first LN

  const dim3 gemm_grid(DDIM / 128, MDIM / 128);  // (8, 64)

  // 1) convert inputs to bf16
  cvt_bf16_kernel<<<(int)(MD / 4 / 256), 256, 0, stream>>>(x, xb, (int)(MD / 4));
  cvt_bf16_kernel<<<(int)(DD / 4 / 256), 256, 0, stream>>>(wq, wqb, (int)(DD / 4));
  cvt_bf16_kernel<<<(int)(DD / 4 / 256), 256, 0, stream>>>(wk, wkb, (int)(DD / 4));
  cvt_bf16_kernel<<<(int)(DD / 4 / 256), 256, 0, stream>>>(wv, wvb, (int)(DD / 4));
  cvt_bf16_kernel<<<(int)(DD / 4 / 256), 256, 0, stream>>>(wo, wob, (int)(DD / 4));
  cvt_bf16_kernel<<<(int)(DD / 4 / 256), 256, 0, stream>>>(wf, wfb, (int)(DD / 4));

  // 2) projections: Q transposed per head, K/V row-major per head
  gemm_bf16_kernel<1><<<gemm_grid, 256, 0, stream>>>(xb, wqb, bq, Qtb, DDIM, DDIM);
  gemm_bf16_kernel<0><<<gemm_grid, 256, 0, stream>>>(xb, wkb, bk, Kb, DDIM, DDIM);
  gemm_bf16_kernel<0><<<gemm_grid, 256, 0, stream>>>(xb, wvb, bv, Vb, DDIM, DDIM);

  // 3) flash attention -> ctx [B,S,D] bf16
  flash_attn_kernel<<<1024, 256, 0, stream>>>(Qtb, Kb, Vb, ctxb);

  // 4) output projection -> y (f32)
  gemm_bf16_kernel<3><<<gemm_grid, 256, 0, stream>>>(ctxb, wob, bo, y, DDIM, DDIM);

  // 5) n1 = LN(x + y)  (also bf16 copy for FFN GEMM)
  add_layernorm_kernel<true><<<MDIM, 256, 0, stream>>>(x, y, g1, b1, n1, n1b);

  // 6) fnn = relu(n1 @ wf + bf)  (f32)
  gemm_bf16_kernel<4><<<gemm_grid, 256, 0, stream>>>(n1b, wfb, bff, fn, DDIM, DDIM);

  // 7) out = LN(fnn + n1)
  add_layernorm_kernel<false><<<MDIM, 256, 0, stream>>>(fn, n1, g2, b2, out,
                                                        nullptr);
}
